// MultiHeadDotProductAttention_65996467470950
// MI455X (gfx1250) — matrix operand order repaired and checked by reading, not compiled, not run on hardware
//
#include <hip/hip_runtime.h>
#include <hip/hip_bf16.h>

// ---------------------------------------------------------------------------
// Problem constants (reference: B=2, L=2048, E=2048, H=16, D=128, fp32 in/out)
// ---------------------------------------------------------------------------
#define B_ 2
#define L_ 2048
#define E_ 2048
#define H_ 16
#define D_ 128
#define LSMAX 4.60517018598809136804f   // log(1/0.01)

#if defined(__has_builtin)
#if __has_builtin(__builtin_amdgcn_tensor_load_to_lds)
#define USE_TDM 1
#endif
#endif
#ifndef USE_TDM
#define USE_TDM 0
#endif

// Compile-probe (non-destructive): report whether the TDM builtin is live.
#if USE_TDM
#warning "CDNA5 probe: USE_TDM=1 (tensor_load_to_lds path compiled in)"
#else
#warning "CDNA5 probe: USE_TDM=0 (tensor_load_to_lds builtin unavailable)"
#endif
#if __clang_major__ >= 23
#warning "CDNA5 probe: clang >= 23 (6-arg TDM builtin form)"
#else
#warning "CDNA5 probe: clang < 23 (5-arg TDM builtin form)"
#endif

typedef __attribute__((ext_vector_type(16))) __bf16 v16bf;
typedef __attribute__((ext_vector_type(8)))  float  v8f;
typedef __attribute__((ext_vector_type(4)))  unsigned int v4u;
typedef __attribute__((ext_vector_type(8)))  int    v8i;
typedef __attribute__((ext_vector_type(4)))  int    v4i;

static __device__ __forceinline__ unsigned int f2bf1(float f) {
    unsigned int u = __float_as_uint(f);
    return (u + 0x7FFFu + ((u >> 16) & 1u)) >> 16;   // round-to-nearest-even
}
static __device__ __forceinline__ unsigned int pack2bf(float a, float b) {
    return f2bf1(a) | (f2bf1(b) << 16);
}

union FragU { uint4 u[2]; v16bf v; };
static __device__ __forceinline__ v16bf mkfrag(uint4 lo, uint4 hi) {
    FragU t; t.u[0] = lo; t.u[1] = hi; return t.v;
}
union U4S8 { uint4 u; unsigned short s[8]; };

// ---------------------------------------------------------------------------
// GEMM: C[M,N] = A[M,K] @ W[N,K]^T   (fp32 C; A fp32 or bf16; W fp32)
// 256 threads = 8 waves, block tile 128(M) x 256(N), K-tile 64.
// Wave grid 2x4: each wave computes 64x64 via 4x4 WMMA 16x16 accumulators
// (16 ds_load_b128 feed 16 WMMAs per 32-K step -> 1.0 loads/WMMA).
// ---------------------------------------------------------------------------
template <bool A_BF16>
__global__ __launch_bounds__(256) void gemm_wmma_bf16(
    const void* __restrict__ Aopaque, const float* __restrict__ W,
    float* __restrict__ C, int M, int N, int K)
{
    __shared__ alignas(16) unsigned short As[128 * 72];
    __shared__ alignas(16) unsigned short Bs[256 * 72];

    const int tid   = threadIdx.x;
    const int lane  = tid & 31;
    const int wave  = tid >> 5;
    const int l15   = lane & 15;
    const int lhalf = lane >> 4;
    const int wm    = wave & 1;   // 0..1 -> 64-row slab
    const int wn    = wave >> 1;  // 0..3 -> 64-col slab
    const int bm    = blockIdx.y * 128;
    const int bn    = blockIdx.x * 256;

    const float*          Af = (const float*)Aopaque;
    const unsigned short* Ab = (const unsigned short*)Aopaque;

    const v8f vz = {0.f, 0.f, 0.f, 0.f, 0.f, 0.f, 0.f, 0.f};
    v8f acc[4][4];
#pragma unroll
    for (int i = 0; i < 4; ++i)
#pragma unroll
        for (int j = 0; j < 4; ++j) acc[i][j] = vz;

    for (int k0 = 0; k0 < K; k0 += 64) {
        // ---- stage A tile (128 x 64) to LDS as bf16 ----
        if (!A_BF16) {
#pragma unroll
            for (int i = 0; i < 8; ++i) {
                int id = tid + i * 256;
                int r = id >> 4, c = (id & 15) * 4;
                const float4 a4 = *(const float4*)(Af + (size_t)(bm + r) * K + k0 + c);
                uint2 p; p.x = pack2bf(a4.x, a4.y); p.y = pack2bf(a4.z, a4.w);
                *(uint2*)&As[r * 72 + c] = p;
            }
        } else {
#pragma unroll
            for (int i = 0; i < 4; ++i) {
                int id = tid + i * 256;
                int r = id >> 3, c = (id & 7) * 8;
                *(uint4*)&As[r * 72 + c] =
                    *(const uint4*)(Ab + (size_t)(bm + r) * K + k0 + c);
            }
        }
        // ---- stage W tile (256 rows of W, 64 k) ----
#pragma unroll
        for (int i = 0; i < 16; ++i) {
            int id = tid + i * 256;
            int r = id >> 4, c = (id & 15) * 4;
            const float4 w4 = *(const float4*)(W + (size_t)(bn + r) * K + k0 + c);
            uint2 p; p.x = pack2bf(w4.x, w4.y); p.y = pack2bf(w4.z, w4.w);
            *(uint2*)&Bs[r * 72 + c] = p;
        }
        if (k0 + 64 < K) {   // global_prefetch of next tiles
            __builtin_prefetch((const char*)Aopaque +
                               ((size_t)(bm + (tid >> 1)) * K + k0 + 64) * (A_BF16 ? 2 : 4),
                               0, 1);
            __builtin_prefetch(W + (size_t)(bn + tid) * K + k0 + 64, 0, 1);
        }
        __syncthreads();

#pragma unroll
        for (int kc = 0; kc < 64; kc += 32) {
            v16bf af[4], bf[4];
#pragma unroll
            for (int i = 0; i < 4; ++i) {
                // A layout: elems 0-7 <- K = kc + lhalf*8 + [0,8); elems 8-15 <- +16
                const unsigned short* ar =
                    &As[(wm * 64 + i * 16 + l15) * 72 + kc + lhalf * 8];
                af[i] = mkfrag(*(const uint4*)ar, *(const uint4*)(ar + 16));
            }
#pragma unroll
            for (int j = 0; j < 4; ++j) {
                // B layout: elems 0-7 <- K = kc + lhalf*8 + [0,8); elems 8-15 <- +16
                const unsigned short* br =
                    &Bs[(wn * 64 + j * 16 + l15) * 72 + kc + lhalf * 8];
                bf[j] = mkfrag(*(const uint4*)br, *(const uint4*)(br + 16));
            }
#pragma unroll
            for (int i = 0; i < 4; ++i)
#pragma unroll
                for (int j = 0; j < 4; ++j)
                    acc[i][j] = __builtin_amdgcn_wmma_f32_16x16x32_bf16(
                        false, af[i], false, bf[j], (short)0, acc[i][j], false, false);
        }
        __syncthreads();
    }

    // C layout: lanes 0-15 -> N=lane, M=e; lanes 16-31 -> N=lane-16, M=e+8
#pragma unroll
    for (int i = 0; i < 4; ++i)
#pragma unroll
        for (int j = 0; j < 4; ++j)
#pragma unroll
            for (int e = 0; e < 8; ++e) {
                int m = bm + wm * 64 + i * 16 + lhalf * 8 + e;
                int n = bn + wn * 64 + j * 16 + l15;
                C[(size_t)m * N + n] = acc[i][j][e];
            }
}

// ---------------------------------------------------------------------------
// RMSNorm + RoPE + L2-normalize (+ fold logit scale into q) -> bf16 heads
// One wave per (b,l,h) row of D=128; lane owns 4 consecutive d.
// Output layout: [B, H, L, D] bf16.
// ---------------------------------------------------------------------------
static __device__ __forceinline__ void norm_rope_l2(
    const float* __restrict__ raw, const float* __restrict__ rope,
    const float* __restrict__ nscale, float ls, unsigned short* __restrict__ out)
{
    float4 x = *(const float4*)raw;
    float ss = x.x * x.x + x.y * x.y + x.z * x.z + x.w * x.w;
#pragma unroll
    for (int off = 16; off > 0; off >>= 1) ss += __shfl_xor(ss, off, 32);
    float inv = rsqrtf(ss * (1.0f / (float)D_) + 1e-6f);
    float4 g = *(const float4*)nscale;
    float y0 = g.x * x.x * inv, y1 = g.y * x.y * inv;
    float y2 = g.z * x.z * inv, y3 = g.w * x.w * inv;
    float4 r = *(const float4*)rope;   // (c0,s0,c1,s1)
    float o0 = y0 * r.x - y1 * r.y;
    float o1 = y1 * r.x + y0 * r.y;
    float o2 = y2 * r.z - y3 * r.w;
    float o3 = y3 * r.z + y2 * r.w;
    float n2 = o0 * o0 + o1 * o1 + o2 * o2 + o3 * o3;
#pragma unroll
    for (int off = 16; off > 0; off >>= 1) n2 += __shfl_xor(n2, off, 32);
    float s = ls / fmaxf(sqrtf(n2), 1e-12f);
    uint2 p; p.x = pack2bf(o0 * s, o1 * s); p.y = pack2bf(o2 * s, o3 * s);
    *(uint2*)out = p;
}

__global__ __launch_bounds__(256) void qkv_prep(
    const float* __restrict__ Qraw, const float* __restrict__ Kraw,
    const float* __restrict__ Vraw,
    const float* __restrict__ ropeQ, const float* __restrict__ ropeK,
    const float* __restrict__ qscale, const float* __restrict__ kscale,
    const float* __restrict__ logit_scale,
    unsigned short* __restrict__ Qh, unsigned short* __restrict__ Kh,
    unsigned short* __restrict__ Vh)
{
    int gw   = blockIdx.x * 8 + (threadIdx.x >> 5);   // row over B*L*H
    int lane = threadIdx.x & 31;
    int h  = gw % H_;
    int bl = gw / H_;
    int b  = bl / L_, l = bl % L_;
    int d0 = lane * 4;
    size_t raw  = (size_t)bl * E_ + (size_t)h * D_ + d0;
    size_t head = (((size_t)b * H_ + h) * L_ + l) * D_ + d0;
    size_t rp   = (size_t)bl * D_ + d0;
    float ls = __expf(fminf(logit_scale[h], LSMAX));
    norm_rope_l2(Qraw + raw, ropeQ + rp, qscale + d0, ls,   Qh + head);
    norm_rope_l2(Kraw + raw, ropeK + rp, kscale + d0, 1.0f, Kh + head);
    float4 v = *(const float4*)(Vraw + raw);
    uint2 p; p.x = pack2bf(v.x, v.y); p.y = pack2bf(v.z, v.w);
    *(uint2*)(Vh + head) = p;
}

// ---------------------------------------------------------------------------
// Flash attention: grid = (L/128, B*H), 256 threads (8 waves).
// Each wave owns 16 query rows; loops over 64-key tiles staged in LDS.
// K tile staged by the Tensor Data Mover (tensor_load_to_lds + TENSORcnt)
// with per-row LDS padding matching the 136-element bank-conflict stride.
// Scores (16x64) via 4x(4 WMMA K=32) with double-buffered K fragments so
// ds_load_b128s stay a full K-step ahead of the WMMAs; online softmax via
// half-wave shfl_xor; P staged per-wave in LDS; P@V via WMMA against V^T.
// ---------------------------------------------------------------------------
__global__ __launch_bounds__(256) void attn_fa(
    const unsigned short* __restrict__ Qh,
    const unsigned short* __restrict__ Kh,
    const unsigned short* __restrict__ Vh,
    const float* __restrict__ bias,
    unsigned short* __restrict__ AOut)
{
    __shared__ alignas(16) unsigned short Ks[64 * 136];    // K tile  [key][d]
    __shared__ alignas(16) unsigned short Vts[128 * 72];   // V^T     [d][key]
    __shared__ alignas(16) unsigned short Ps[8][16 * 72];  // per-wave P tile

    const int tid   = threadIdx.x;
    const int lane  = tid & 31;
    const int wave  = tid >> 5;
    const int l15   = lane & 15;
    const int lhalf = lane >> 4;
    const int bh = blockIdx.y, b = bh / H_, h = bh % H_;
    const int q0 = blockIdx.x * 128 + wave * 16;

    const unsigned short* Qb = Qh + (size_t)bh * L_ * D_;
    const unsigned short* Kb = Kh + (size_t)bh * L_ * D_;
    const unsigned short* Vb = Vh + (size_t)bh * L_ * D_;

#if USE_TDM
    // Make LDS stores to Ks visible to the compiler (TDM writes are opaque).
    {
        const uint4 z4 = {0u, 0u, 0u, 0u};
        for (int i = tid; i < (64 * 136) / 8; i += 256) ((uint4*)Ks)[i] = z4;
    }
    const unsigned int ldsK =
        __builtin_amdgcn_readfirstlane((unsigned int)(size_t)(const void*)&Ks[0]);
#endif

    // Q fragments held in registers for the whole kernel (16 rows x 128 d)
    v16bf qf[4];
    {
        const unsigned short* qr = Qb + (size_t)(q0 + l15) * D_;
#pragma unroll
        for (int kc = 0; kc < 4; ++kc) {
            const unsigned short* p = qr + kc * 32 + lhalf * 16;
            qf[kc] = mkfrag(*(const uint4*)p, *(const uint4*)(p + 8));
        }
    }

    const v8f vz = {0.f, 0.f, 0.f, 0.f, 0.f, 0.f, 0.f, 0.f};
    float mrow[8], lrow[8];
    v8f oacc[8];
#pragma unroll
    for (int e = 0; e < 8; ++e) { mrow[e] = -3.0e38f; lrow[e] = 0.f; }
#pragma unroll
    for (int n = 0; n < 8; ++n) oacc[n] = vz;

    for (int kt = 0; kt < L_ / 64; ++kt) {
        const int k0g = kt * 64;
        __syncthreads();

#if USE_TDM
        // ---- TDM: DMA the 64x128 bf16 K tile into LDS (wave 0 issues once) --
        if (wave == 0) {
            unsigned long long ga = (unsigned long long)(size_t)(Kb + (size_t)k0g * D_);
            unsigned int galo = __builtin_amdgcn_readfirstlane((unsigned int)ga);
            unsigned int gahi = __builtin_amdgcn_readfirstlane((unsigned int)(ga >> 32));
            // D# group0: count=1 | lds_addr | global_addr | type=2 ("image")
            v4u g0 = { 1u, ldsK, galo, gahi | (2u << 30) };
            // D# group1: data_size=2B, pad every 64 DWORDs by 4 DWORDs
            // (256B row + 16B pad == 136-ushort LDS stride), tile 128x64,
            // tensor_dim0=stride0=128 elements, tensor_dim1=L.
            unsigned int w0 = (1u << 16) | (1u << 20) | (5u << 22) | (3u << 25);
            v8i g1 = { (int)w0,
                       (int)((unsigned int)D_ << 16),   // tensor_dim0[15:0] @ bits 63:48
                       (int)((unsigned int)L_ << 16),   // tensor_dim1[15:0] @ bits 95:80
                       (int)((unsigned int)D_ << 16),   // tile_dim0        @ bits 127:112
                       64,                              // tile_dim1 = 64 rows
                       D_,                              // tensor_dim0_stride[31:0]
                       0, 0 };
            v4i gz = { 0, 0, 0, 0 };
#if __clang_major__ >= 23
            v8i gz8 = { 0, 0, 0, 0, 0, 0, 0, 0 };
            __builtin_amdgcn_tensor_load_to_lds(g0, g1, gz, gz, gz8, 0);
#else
            __builtin_amdgcn_tensor_load_to_lds(g0, g1, gz, gz, 0);
#endif
        }
#else
        // ---- stage K tile (64 x 128 bf16) manually ----
#pragma unroll
        for (int i = 0; i < 4; ++i) {
            int id = tid + i * 256;
            int r = id >> 4, c = (id & 15) * 8;
            *(uint4*)&Ks[r * 136 + c] =
                *(const uint4*)(Kb + (size_t)(k0g + r) * D_ + c);
        }
#endif
        // ---- stage V transposed (d-major) for contiguous B fragments ----
#pragma unroll
        for (int i = 0; i < 2; ++i) {
            int id = tid + i * 256;
            int r = id >> 4, c = (id & 15) * 8;
            U4S8 u; u.u = *(const uint4*)(Vb + (size_t)(k0g + r) * D_ + c);
#pragma unroll
            for (int j = 0; j < 8; ++j) Vts[(c + j) * 72 + r] = u.s[j];
        }
#if USE_TDM
        if (wave == 0) __builtin_amdgcn_s_wait_tensorcnt(0);
#endif
        __syncthreads();

        // ---- scores: 16 x 64; K fragments double-buffered one K-step ahead
        v8f s[4];
#pragma unroll
        for (int j = 0; j < 4; ++j) s[j] = vz;
        v16bf kf[2][4];
#pragma unroll
        for (int j = 0; j < 4; ++j) {
            const unsigned short* p = &Ks[(j * 16 + l15) * 136 + lhalf * 16];
            kf[0][j] = mkfrag(*(const uint4*)p, *(const uint4*)(p + 8));
        }
#pragma unroll
        for (int kc = 0; kc < 4; ++kc) {
            if (kc < 3) {
#pragma unroll
                for (int j = 0; j < 4; ++j) {
                    const unsigned short* p =
                        &Ks[(j * 16 + l15) * 136 + (kc + 1) * 32 + lhalf * 16];
                    kf[(kc + 1) & 1][j] = mkfrag(*(const uint4*)p, *(const uint4*)(p + 8));
                }
            }
#pragma unroll
            for (int j = 0; j < 4; ++j)
                s[j] = __builtin_amdgcn_wmma_f32_16x16x32_bf16(
                    false, qf[kc], false, kf[kc & 1][j], (short)0, s[j], false, false);
        }
        // ---- bias add (logit scale already folded into q̂)
#pragma unroll
        for (int j = 0; j < 4; ++j)
#pragma unroll
            for (int e = 0; e < 8; ++e)
                s[j][e] += bias[(size_t)(q0 + lhalf * 8 + e) * L_ +
                                (k0g + j * 16 + l15)];

        // ---- online softmax (row reductions within each 16-lane half)
        float mt[8];
#pragma unroll
        for (int e = 0; e < 8; ++e)
            mt[e] = fmaxf(fmaxf(s[0][e], s[1][e]), fmaxf(s[2][e], s[3][e]));
#pragma unroll
        for (int off = 8; off > 0; off >>= 1)
#pragma unroll
            for (int e = 0; e < 8; ++e)
                mt[e] = fmaxf(mt[e], __shfl_xor(mt[e], off, 32));

        float corr[8], rs[8];
#pragma unroll
        for (int e = 0; e < 8; ++e) {
            float mn = fmaxf(mrow[e], mt[e]);
            corr[e] = __expf(mrow[e] - mn);
            mrow[e] = mn;
            rs[e] = 0.f;
        }
#pragma unroll
        for (int j = 0; j < 4; ++j)
#pragma unroll
            for (int e = 0; e < 8; ++e) {
                float p = __expf(s[j][e] - mrow[e]);
                s[j][e] = p;
                rs[e] += p;
            }
#pragma unroll
        for (int off = 8; off > 0; off >>= 1)
#pragma unroll
            for (int e = 0; e < 8; ++e)
                rs[e] += __shfl_xor(rs[e], off, 32);
#pragma unroll
        for (int e = 0; e < 8; ++e) lrow[e] = lrow[e] * corr[e] + rs[e];
#pragma unroll
        for (int n = 0; n < 8; ++n)
#pragma unroll
            for (int e = 0; e < 8; ++e)
                oacc[n][e] *= corr[e];

        // ---- P -> per-wave LDS (bf16); same-wave DS ops are in-order
        unsigned short* pw = Ps[wave];
#pragma unroll
        for (int j = 0; j < 4; ++j)
#pragma unroll
            for (int e = 0; e < 8; ++e)
                pw[(lhalf * 8 + e) * 72 + j * 16 + l15] =
                    (unsigned short)f2bf1(s[j][e]);

        // ---- O += P @ V  (keys reduced 32 at a time)
#pragma unroll
        for (int kc2 = 0; kc2 < 2; ++kc2) {
            const unsigned short* pr = &pw[l15 * 72 + kc2 * 32 + lhalf * 16];
            v16bf pa = mkfrag(*(const uint4*)pr, *(const uint4*)(pr + 8));
#pragma unroll
            for (int n = 0; n < 8; ++n) {
                const unsigned short* vr =
                    &Vts[(n * 16 + l15) * 72 + kc2 * 32 + lhalf * 16];
                v16bf vfb = mkfrag(*(const uint4*)vr, *(const uint4*)(vr + 8));
                oacc[n] = __builtin_amdgcn_wmma_f32_16x16x32_bf16(
                    false, pa, false, vfb, (short)0, oacc[n], false, false);
            }
        }
    }

    // ---- normalize and store attention output as bf16 [B, L, H*D]
    float invl[8];
#pragma unroll
    for (int e = 0; e < 8; ++e) invl[e] = 1.0f / lrow[e];
#pragma unroll
    for (int n = 0; n < 8; ++n)
#pragma unroll
        for (int e = 0; e < 8; ++e) {
            int qrow = q0 + lhalf * 8 + e;
            int d = n * 16 + l15;
            AOut[(((size_t)b * L_ + qrow) * H_ + h) * D_ + d] =
                (unsigned short)f2bf1(oacc[n][e] * invl[e]);
        }
}

// ---------------------------------------------------------------------------
// Host-side orchestration (graph-capture safe: launches only).
// ---------------------------------------------------------------------------
extern "C" void kernel_launch(void* const* d_in, const int* in_sizes, int n_in,
                              void* d_out, int out_size, void* d_ws, size_t ws_size,
                              hipStream_t stream)
{
    (void)in_sizes; (void)n_in; (void)out_size; (void)ws_size;
    const float* inputs_q  = (const float*)d_in[0];
    const float* inputs_kv = (const float*)d_in[1];
    const float* bias      = (const float*)d_in[2];
    const float* q_sin     = (const float*)d_in[3];
    const float* k_sin     = (const float*)d_in[4];
    const float* Wq        = (const float*)d_in[5];
    const float* Wk        = (const float*)d_in[6];
    const float* Wv        = (const float*)d_in[7];
    const float* Wo        = (const float*)d_in[8];
    const float* qns       = (const float*)d_in[9];
    const float* kns       = (const float*)d_in[10];
    const float* lsc       = (const float*)d_in[11];

    char* ws = (char*)d_ws;
    const size_t SRAW = (size_t)B_ * L_ * E_ * sizeof(float);          // 33.5 MB
    const size_t SH   = (size_t)B_ * L_ * E_ * sizeof(unsigned short); // 16.8 MB
    float* Qraw = (float*)(ws);
    float* Kraw = (float*)(ws + SRAW);
    float* Vraw = (float*)(ws + 2 * SRAW);
    unsigned short* Qh = (unsigned short*)(ws + 3 * SRAW);
    unsigned short* Kh = (unsigned short*)(ws + 3 * SRAW + SH);
    unsigned short* Vh = (unsigned short*)(ws + 3 * SRAW + 2 * SH);
    unsigned short* AOut = (unsigned short*)(ws);   // alias Qraw (dead after prep)

    const dim3 gGemm(E_ / 256, (B_ * L_) / 128);    // (8, 32)
    gemm_wmma_bf16<false><<<gGemm, 256, 0, stream>>>(inputs_q,  Wq, Qraw, B_ * L_, E_, E_);
    gemm_wmma_bf16<false><<<gGemm, 256, 0, stream>>>(inputs_kv, Wk, Kraw, B_ * L_, E_, E_);
    gemm_wmma_bf16<false><<<gGemm, 256, 0, stream>>>(inputs_kv, Wv, Vraw, B_ * L_, E_, E_);

    qkv_prep<<<(B_ * L_ * H_) / 8, 256, 0, stream>>>(
        Qraw, Kraw, Vraw, q_sin, k_sin, qns, kns, lsc, Qh, Kh, Vh);

    attn_fa<<<dim3(L_ / 128, B_ * H_), 256, 0, stream>>>(Qh, Kh, Vh, bias, AOut);

    gemm_wmma_bf16<true><<<gGemm, 256, 0, stream>>>(AOut, Wo, (float*)d_out, B_ * L_, E_, E_);
}
